// PyG_MPNNGNN_2010044694732
// MI455X (gfx1250) — compile-verified, matrix operand order checked by reading, not run on hardware
//
#include <hip/hip_runtime.h>
#include <hip/hip_bf16.h>
#include <stdint.h>

#define N_NODES 10000
#define N_EDGES 100000
#define D 64
#define EHID 128
#define NI 74
#define EI 12
#define STEPS 6
#define KP_X 96      /* NI padded to mult of 32 */
#define KP_EA 32     /* EI padded to 32 */
#define GW 8256      /* 128*64 + 64 bias columns */
#define G3 192       /* 3*D gru gates */

typedef __attribute__((ext_vector_type(16))) __bf16 bf16x16;
typedef __attribute__((ext_vector_type(8)))  __bf16 bf16x8;
typedef __attribute__((ext_vector_type(8)))  float  f32x8;

// ---------------------------------------------------------------------------
// WMMA bf16 fragment mapping per CDNA5 ISA 7.12.2 (wave32), 16x16x32:
//   frag element j of lane l corresponds to k = (j&7) + 8*half + 16*(j>>3),
//   half = l>>4;  A: row = l&15;  B: col = l&15.
// B matrices are PRE-SWIZZLED so each lane's 16 elements are contiguous:
//   offset((ntile,kchunk),lane,j) = (((ntile*Kc + kchunk)*32) + lane)*16 + j
// -> one aligned 32-byte load per fragment instead of 16 u16 gathers.
// ---------------------------------------------------------------------------
__device__ __forceinline__ size_t bsw_off(int k, int col, int K, int Nout) {
  const int kc = k >> 5, kr = k & 31;
  const int nt = col >> 4, c = col & 15;
  const int half = (kr >> 3) & 1;
  const int j = (kr & 7) + ((kr >> 4) << 3);
  const int lane = c + (half << 4);
  (void)Nout;
  return (((size_t)nt * (K >> 5) + kc) * 32 + lane) * 16 + j;
}

__device__ __forceinline__ bf16x16 load_a_frag(const __bf16* __restrict__ A,
                                               int lda, int lane) {
  const int half = lane >> 4, r = lane & 15;
  const __bf16* row = A + (size_t)r * lda + 8 * half;
  bf16x8 lo = *reinterpret_cast<const bf16x8*>(row);
  bf16x8 hi = *reinterpret_cast<const bf16x8*>(row + 16);
  bf16x16 f;
#pragma unroll
  for (int j = 0; j < 8; ++j) { f[j] = lo[j]; f[8 + j] = hi[j]; }
  return f;
}

// ---------------------------------------------------------------------------
// Generic bf16 WMMA GEMM: C[M,Nout] = act(A[M,K] @ B[K,Nout] + bias)
// A row-major bf16; B pre-swizzled fragment layout (see above); one 16x16
// tile per wave, NT consecutive n-tiles per wave to amortize the A fragment.
// ---------------------------------------------------------------------------
template <int NT, bool RELU>
__global__ __launch_bounds__(256) void wmma_gemm_kernel(
    const __bf16* __restrict__ A, const __bf16* __restrict__ Bsw,
    const float* __restrict__ bias, float* __restrict__ Cf,
    __bf16* __restrict__ Cb, int M, int Nout, int K)
{
  const int lane = threadIdx.x & 31;
  const int wid  = blockIdx.x * (blockDim.x >> 5) + (threadIdx.x >> 5);
  const int ntG  = (Nout >> 4) / NT;
  const int mt   = wid / ntG;
  const int ng   = wid % ntG;
  if (mt * 16 >= M) return;
  const int half = lane >> 4;
  const int rc   = lane & 15;
  const int kchunks = K >> 5;

  f32x8 acc[NT];
#pragma unroll
  for (int t = 0; t < NT; ++t) acc[t] = {};

  for (int kc = 0; kc < kchunks; ++kc) {
    bf16x16 afrag = load_a_frag(A + (size_t)mt * 16 * K + kc * 32, K, lane);
#pragma unroll
    for (int t = 0; t < NT; ++t) {
      const size_t blk = (size_t)(ng * NT + t) * kchunks + kc;
      bf16x16 bfrag =
          *reinterpret_cast<const bf16x16*>(Bsw + (blk * 32 + lane) * 16);
      acc[t] = __builtin_amdgcn_wmma_f32_16x16x32_bf16(
          false, afrag, false, bfrag, (short)0, acc[t], false, false);
    }
  }

#pragma unroll
  for (int t = 0; t < NT; ++t) {
    const int col = (ng * NT + t) * 16 + rc;
    const float bv = bias ? bias[col] : 0.0f;
#pragma unroll
    for (int i = 0; i < 8; ++i) {
      const int row = mt * 16 + i + 8 * half;
      float v = acc[t][i] + bv;
      if (RELU) v = fmaxf(v, 0.0f);
      const size_t o = (size_t)row * Nout + col;
      if (Cf) Cf[o] = v;
      if (Cb) Cb[o] = (__bf16)v;
    }
  }
}

// ---------------------------------------------------------------------------
// Per-edge message + scatter-add.  One wave per edge; lane covers features
// {2*lane, 2*lane+1}.  msg[e,f] = sum_h eh[e,h]*G[src,h*64+f] + G[src,8192+f]
// G packed bf16 pairs -> one b32 load per lane per h (128B coalesced, L2 hit).
// eh[e,h] broadcast from 4 preloaded regs via v_readlane.
// ---------------------------------------------------------------------------
__global__ __launch_bounds__(256) void edge_msg_kernel(
    const int* __restrict__ ei, const __bf16* __restrict__ eh,
    const __bf16* __restrict__ G, float* __restrict__ agg)
{
  const int lane = threadIdx.x & 31;
  const int e = blockIdx.x * (blockDim.x >> 5) + (threadIdx.x >> 5);
  if (e >= N_EDGES) return;
  const int s = ei[e];
  const int d = ei[N_EDGES + e];
  const unsigned int* __restrict__ Gp =
      reinterpret_cast<const unsigned int*>(G + (size_t)s * GW);
  __builtin_prefetch(Gp + lane, 0, 1);  // global_prefetch_b8 on the G row

  float ehv[4];
#pragma unroll
  for (int i = 0; i < 4; ++i)
    ehv[i] = (float)eh[(size_t)e * EHID + i * 32 + lane];

  float a0 = 0.0f, a1 = 0.0f;
#pragma unroll
  for (int i = 0; i < 4; ++i) {
    const unsigned int esrc = __float_as_uint(ehv[i]);
#pragma unroll
    for (int hh = 0; hh < 32; ++hh) {
      const float ev = __uint_as_float(__builtin_amdgcn_readlane(esrc, hh));
      const unsigned int g = Gp[(i * 32 + hh) * 32 + lane];
      a0 = fmaf(ev, __uint_as_float(g << 16), a0);
      a1 = fmaf(ev, __uint_as_float(g & 0xffff0000u), a1);
    }
  }
  // xp @ b_e2 bias columns (cols 8192..8255 of G)
  const unsigned int gb = Gp[128 * 32 + lane];
  a0 += __uint_as_float(gb << 16);
  a1 += __uint_as_float(gb & 0xffff0000u);

  float* ap = agg + (size_t)d * D + 2 * lane;
  __hip_atomic_fetch_add(ap,     a0, __ATOMIC_RELAXED, __HIP_MEMORY_SCOPE_AGENT);
  __hip_atomic_fetch_add(ap + 1, a1, __ATOMIC_RELAXED, __HIP_MEMORY_SCOPE_AGENT);
}

// ------------------------------- elementwise -------------------------------
__global__ void k_zero_f32(float* p, int n) {
  int i = blockIdx.x * blockDim.x + threadIdx.x;
  if (i < n) p[i] = 0.0f;
}
__global__ void k_copy_bf16(const __bf16* s, __bf16* d, int n) {
  int i = blockIdx.x * blockDim.x + threadIdx.x;
  if (i < n) d[i] = s[i];
}
// rows of fp32 -> bf16, K padded with zeros (A-side, stays row-major)
__global__ void k_pad_rows(const float* __restrict__ in, __bf16* __restrict__ o,
                           int rows, int kin, int kpad) {
  int i = blockIdx.x * blockDim.x + threadIdx.x;
  if (i >= rows * kpad) return;
  int r = i / kpad, k = i % kpad;
  o[i] = (k < kin) ? (__bf16)in[(size_t)r * kin + k] : (__bf16)0.0f;
}
// W[nout,kin] fp32 -> swizzled-fragment B [kpad x nout] bf16
__global__ void k_transpose_pad_sw(const float* __restrict__ W, __bf16* __restrict__ T,
                                   int kin, int kpad, int nout) {
  int i = blockIdx.x * blockDim.x + threadIdx.x;
  if (i >= kpad * nout) return;
  int k = i / nout, n = i % nout;
  float v = (k < kin) ? W[(size_t)n * kin + k] : 0.0f;
  T[bsw_off(k, n, kpad, nout)] = (__bf16)v;
}
// W2aug[k=d][col=h*64+f] = W_e2[(d*64+f), h] ; col 8192+f -> b_e2[d*64+f]
// stored directly in swizzled-fragment layout (K=64, Nout=GW)
__global__ void k_w2aug_sw(const float* __restrict__ W_e2, const float* __restrict__ b_e2,
                           __bf16* __restrict__ T) {
  int i = blockIdx.x * blockDim.x + threadIdx.x;
  if (i >= D * GW) return;
  int d = i / GW, c = i % GW;
  float v;
  if (c < 8192) {
    int h = c >> 6, f = c & 63;
    v = W_e2[(size_t)(d * 64 + f) * EHID + h];
  } else {
    v = b_e2[d * 64 + (c - 8192)];
  }
  T[bsw_off(d, c, D, GW)] = (__bf16)v;
}
__global__ void k_count(const int* __restrict__ ei, float* __restrict__ cnt) {
  int e = blockIdx.x * blockDim.x + threadIdx.x;
  if (e >= N_EDGES) return;
  __hip_atomic_fetch_add(&cnt[ei[N_EDGES + e]], 1.0f,
                         __ATOMIC_RELAXED, __HIP_MEMORY_SCOPE_AGENT);
}
// nf = bf16(relu(agg / max(cnt,1)))
__global__ void k_nf(const float* __restrict__ agg, const float* __restrict__ cnt,
                     __bf16* __restrict__ nfb, int n) {
  int i = blockIdx.x * blockDim.x + threadIdx.x;
  if (i >= n) return;
  float c = cnt[i >> 6];
  c = (c < 1.0f) ? 1.0f : c;
  float v = agg[i] / c;
  nfb[i] = (__bf16)(v > 0.0f ? v : 0.0f);
}
// GRU gates (torch layout [r;z;n]); hidden is the fixed initial projection
__global__ void k_gate(const float* __restrict__ gi, const float* __restrict__ gh,
                       const float* __restrict__ hidden, float* __restrict__ outf,
                       __bf16* __restrict__ outb, int n) {
  int i = blockIdx.x * blockDim.x + threadIdx.x;
  if (i >= n) return;
  int r = i >> 6, f = i & 63;
  const float ir = gi[(size_t)r * G3 + f];
  const float iz = gi[(size_t)r * G3 + 64 + f];
  const float in_ = gi[(size_t)r * G3 + 128 + f];
  const float hr = gh[(size_t)r * G3 + f];
  const float hz = gh[(size_t)r * G3 + 64 + f];
  const float hn = gh[(size_t)r * G3 + 128 + f];
  const float rg = 1.0f / (1.0f + __expf(-(ir + hr)));
  const float zg = 1.0f / (1.0f + __expf(-(iz + hz)));
  const float ng = tanhf(in_ + rg * hn);
  const float h = hidden[i];
  const float o = (1.0f - zg) * ng + zg * h;
  outf[i] = o;
  outb[i] = (__bf16)o;
}

// ---------------------------------------------------------------------------
extern "C" void kernel_launch(void* const* d_in, const int* in_sizes, int n_in,
                              void* d_out, int out_size, void* d_ws, size_t ws_size,
                              hipStream_t stream) {
  const float* x      = (const float*)d_in[0];
  const int*   ei     = (const int*)d_in[1];
  const float* ea     = (const float*)d_in[2];
  const float* W_proj = (const float*)d_in[3];
  const float* b_proj = (const float*)d_in[4];
  const float* W_e1   = (const float*)d_in[5];
  const float* b_e1   = (const float*)d_in[6];
  const float* W_e2   = (const float*)d_in[7];
  const float* b_e2   = (const float*)d_in[8];
  const float* W_lin  = (const float*)d_in[9];
  const float* b_lin  = (const float*)d_in[10];
  const float* W_ih   = (const float*)d_in[11];
  const float* W_hh   = (const float*)d_in[12];
  const float* b_ih   = (const float*)d_in[13];
  const float* b_hh   = (const float*)d_in[14];
  float* out = (float*)d_out;
  (void)in_sizes; (void)n_in; (void)out_size; (void)ws_size;

  char* ws = (char*)d_ws;
  size_t off = 0;
  auto alloc = [&](size_t bytes) -> char* {
    char* p = ws + off;
    off = (off + bytes + 255) & ~((size_t)255);
    return p;
  };
  __bf16* xbf     = (__bf16*)alloc((size_t)N_NODES * KP_X * 2);
  __bf16* eabf    = (__bf16*)alloc((size_t)N_EDGES * KP_EA * 2);
  __bf16* WprojT  = (__bf16*)alloc((size_t)KP_X * D * 2);
  __bf16* We1T    = (__bf16*)alloc((size_t)KP_EA * EHID * 2);
  __bf16* W2aug   = (__bf16*)alloc((size_t)D * GW * 2);
  __bf16* WlinT   = (__bf16*)alloc((size_t)D * D * 2);
  __bf16* WihT    = (__bf16*)alloc((size_t)D * G3 * 2);
  __bf16* WhhT    = (__bf16*)alloc((size_t)D * G3 * 2);
  float*  hidden  = (float*) alloc((size_t)N_NODES * D * 4);
  __bf16* hiddenb = (__bf16*)alloc((size_t)N_NODES * D * 2);
  __bf16* nodefb  = (__bf16*)alloc((size_t)N_NODES * D * 2);
  __bf16* xpb     = (__bf16*)alloc((size_t)N_NODES * D * 2);
  __bf16* ehb     = (__bf16*)alloc((size_t)N_EDGES * EHID * 2);
  __bf16* G       = (__bf16*)alloc((size_t)N_NODES * GW * 2);
  float*  agg     = (float*) alloc((size_t)N_NODES * D * 4);
  __bf16* nfb     = (__bf16*)alloc((size_t)N_NODES * D * 2);
  float*  gi      = (float*) alloc((size_t)N_NODES * G3 * 4);
  float*  gh      = (float*) alloc((size_t)N_NODES * G3 * 4);
  float*  cnt     = (float*) alloc((size_t)N_NODES * 4);

  auto eg = [](long long n) { return dim3((unsigned)((n + 255) / 256)); };
  auto gemm_grid = [](int M, int Nout, int NT) {
    long long waves = (long long)(M / 16) * ((Nout / 16) / NT);
    return dim3((unsigned)((waves * 32 + 255) / 256));
  };

  // ---- one-time prep -------------------------------------------------------
  k_pad_rows<<<eg((long long)N_NODES * KP_X), 256, 0, stream>>>(x, xbf, N_NODES, NI, KP_X);
  k_pad_rows<<<eg((long long)N_EDGES * KP_EA), 256, 0, stream>>>(ea, eabf, N_EDGES, EI, KP_EA);
  k_transpose_pad_sw<<<eg(KP_X * D), 256, 0, stream>>>(W_proj, WprojT, NI, KP_X, D);
  k_transpose_pad_sw<<<eg(KP_EA * EHID), 256, 0, stream>>>(W_e1, We1T, EI, KP_EA, EHID);
  k_transpose_pad_sw<<<eg(D * D), 256, 0, stream>>>(W_lin, WlinT, D, D, D);
  k_transpose_pad_sw<<<eg(D * G3), 256, 0, stream>>>(W_ih, WihT, D, D, G3);
  k_transpose_pad_sw<<<eg(D * G3), 256, 0, stream>>>(W_hh, WhhT, D, D, G3);
  k_w2aug_sw<<<eg((long long)D * GW), 256, 0, stream>>>(W_e2, b_e2, W2aug);
  k_zero_f32<<<eg(N_NODES), 256, 0, stream>>>(cnt, N_NODES);
  k_count<<<eg(N_EDGES), 256, 0, stream>>>(ei, cnt);

  // hidden = relu(x @ W_proj^T + b_proj)   [WMMA]
  wmma_gemm_kernel<1, true><<<gemm_grid(N_NODES, D, 1), 256, 0, stream>>>(
      xbf, WprojT, b_proj, hidden, hiddenb, N_NODES, D, KP_X);
  k_copy_bf16<<<eg((long long)N_NODES * D), 256, 0, stream>>>(hiddenb, nodefb, N_NODES * D);

  // eh = relu(edge_attr @ W_e1^T + b_e1)   [WMMA]
  wmma_gemm_kernel<1, true><<<gemm_grid(N_EDGES, EHID, 1), 256, 0, stream>>>(
      eabf, We1T, b_e1, (float*)nullptr, ehb, N_EDGES, EHID, KP_EA);

  // gh = hidden @ W_hh^T + b_hh  (loop-invariant: hidden never changes) [WMMA]
  wmma_gemm_kernel<1, false><<<gemm_grid(N_NODES, G3, 1), 256, 0, stream>>>(
      hiddenb, WhhT, b_hh, gh, (__bf16*)nullptr, N_NODES, G3, D);

  // ---- message-passing steps ----------------------------------------------
  for (int step = 0; step < STEPS; ++step) {
    // xp = node_feats @ W_lin^T + b_lin   [WMMA]
    wmma_gemm_kernel<1, false><<<gemm_grid(N_NODES, D, 1), 256, 0, stream>>>(
        nodefb, WlinT, b_lin, (float*)nullptr, xpb, N_NODES, D, D);
    // G = xp @ W2aug  [N,64]x[64,8256]   [WMMA, 4 n-tiles/wave]
    wmma_gemm_kernel<4, false><<<gemm_grid(N_NODES, GW, 4), 256, 0, stream>>>(
        xpb, W2aug, (const float*)nullptr, (float*)nullptr, G, N_NODES, GW, D);
    // agg = scatter-add of per-edge messages
    k_zero_f32<<<eg((long long)N_NODES * D), 256, 0, stream>>>(agg, N_NODES * D);
    edge_msg_kernel<<<dim3(N_EDGES / 8), 256, 0, stream>>>(ei, ehb, G, agg);
    // nf = relu(agg / max(cnt,1))
    k_nf<<<eg((long long)N_NODES * D), 256, 0, stream>>>(agg, cnt, nfb, N_NODES * D);
    // gi = nf @ W_ih^T + b_ih   [WMMA]
    wmma_gemm_kernel<1, false><<<gemm_grid(N_NODES, G3, 1), 256, 0, stream>>>(
        nfb, WihT, b_ih, gi, (__bf16*)nullptr, N_NODES, G3, D);
    // GRU gates -> node_feats (fp32 to d_out every step; last step is final)
    k_gate<<<eg((long long)N_NODES * D), 256, 0, stream>>>(
        gi, gh, hidden, out, nodefb, N_NODES * D);
  }
}
